// Attention_61246233641283
// MI455X (gfx1250) — compile-verified, matrix operand order checked by reading, not run on hardware
//
#include <hip/hip_runtime.h>
#include <hip/hip_bf16.h>

typedef _Float16 f16;
typedef __attribute__((ext_vector_type(16))) _Float16 v16h;
typedef __attribute__((ext_vector_type(8)))  float    v8f;

#define B_    2
#define L_    2048
#define D_    1024
#define H_    16
#define DK_   64
#define MTOT  (B_ * L_)     // 4096
#define NQKV  (3 * D_)      // 3072

// ---------------------------------------------------------------------------
// Fragment loaders.
// A-operand (16xK rows of a row-major f16 matrix) and B-operand (columns of B
// == rows of B^T stored row-major) share the same VGPR layout on CDNA5:
//   lane<16 : row = lane,    K = c0+0..7 (v0-3), c0+16..23 (v4-7)
//   lane>=16: row = lane-16, K = c0+8..15,       c0+24..31
// => two contiguous 16-byte loads per fragment.
// ---------------------------------------------------------------------------
__device__ __forceinline__ v16h load_frag(const f16* __restrict__ p, int ld, int c0) {
  const int lane = threadIdx.x & 31;
  const int row  = lane & 15;
  const int half = lane >> 4;
  const f16* r = p + (size_t)row * ld + c0 + half * 8;
  v16h f;
  ((uint4*)&f)[0] = *(const uint4*)(r);
  ((uint4*)&f)[1] = *(const uint4*)(r + 16);
  return f;
}

// Same fragment, sourced from fp32 memory with on-the-fly f32->f16 convert.
__device__ __forceinline__ v16h load_frag_f32(const float* __restrict__ p, int ld, int c0) {
  const int lane = threadIdx.x & 31;
  const int row  = lane & 15;
  const int half = lane >> 4;
  const float* r = p + (size_t)row * ld + c0 + half * 8;
  float4 a0 = ((const float4*)r)[0];
  float4 a1 = ((const float4*)r)[1];
  float4 b0 = ((const float4*)(r + 16))[0];
  float4 b1 = ((const float4*)(r + 16))[1];
  v16h f;
  f[0]  = (f16)a0.x; f[1]  = (f16)a0.y; f[2]  = (f16)a0.z; f[3]  = (f16)a0.w;
  f[4]  = (f16)a1.x; f[5]  = (f16)a1.y; f[6]  = (f16)a1.z; f[7]  = (f16)a1.w;
  f[8]  = (f16)b0.x; f[9]  = (f16)b0.y; f[10] = (f16)b0.z; f[11] = (f16)b0.w;
  f[12] = (f16)b1.x; f[13] = (f16)b1.y; f[14] = (f16)b1.z; f[15] = (f16)b1.w;
  return f;
}

__device__ __forceinline__ v8f wmma16(v16h a, v16h b, v8f c) {
  return __builtin_amdgcn_wmma_f32_16x16x32_f16(false, a, false, b, (short)0, c,
                                                false, false);
}

// ---------------------------------------------------------------------------
// Kernel 0: transpose fp32 [K x N] -> f16 [N x K]  (weights -> B-operand form)
// ---------------------------------------------------------------------------
__global__ void __launch_bounds__(256)
transpose_f32_to_f16(const float* __restrict__ in, f16* __restrict__ out,
                     int K, int N) {
  __shared__ float tile[32][33];
  const int n0 = blockIdx.x * 32, k0 = blockIdx.y * 32;
  const int tx = threadIdx.x, ty = threadIdx.y;
  #pragma unroll
  for (int i = ty; i < 32; i += 8)
    tile[i][tx] = in[(size_t)(k0 + i) * N + n0 + tx];
  __syncthreads();
  #pragma unroll
  for (int i = ty; i < 32; i += 8)
    out[(size_t)(n0 + i) * K + k0 + tx] = (f16)tile[tx][i];
}

// ---------------------------------------------------------------------------
// Kernel 1: QKV projection  Y = X @ Wqkv + b, scattered into
//   q  [B,H,L,DK] f16, pre-scaled by 1/sqrt(DK)=0.125
//   k  [B,H,L,DK] f16
//   vt [B,H,DK,L] f16  (V transposed so P@V B-operand rows are contiguous)
// One wave computes a 16x64 tile; workgroup = 4 waves = 64x64.
// ---------------------------------------------------------------------------
__global__ void __launch_bounds__(128)
qkv_gemm(const float* __restrict__ x, const f16* __restrict__ wT,
         const float* __restrict__ bias,
         f16* __restrict__ q, f16* __restrict__ k, f16* __restrict__ vt) {
  const int wave = threadIdx.x >> 5;
  const int lane = threadIdx.x & 31;
  const int m0 = blockIdx.y * 64 + wave * 16;
  const int n0 = blockIdx.x * 64;

  v8f zero = {0.f, 0.f, 0.f, 0.f, 0.f, 0.f, 0.f, 0.f};
  v8f acc[4] = {zero, zero, zero, zero};

  const float* xr = x + (size_t)m0 * D_;
  for (int kk = 0; kk < D_; kk += 32) {
    v16h a = load_frag_f32(xr, D_, kk);
    #pragma unroll
    for (int t = 0; t < 4; ++t) {
      v16h b = load_frag(wT + (size_t)(n0 + t * 16) * D_, D_, kk);
      acc[t] = wmma16(a, b, acc[t]);
    }
  }

  // Epilogue: C/D layout -> lane holds col n = n0+t*16+(lane&15),
  //           rows m = m0 + g + 8*(lane>>4).  192 cols per head, 64 | 192 so
  //           head h and Q/K/V selector are uniform per block.
  const int half  = lane >> 4;
  const int which = (n0 % 192) / 64;   // 0=Q 1=K 2=V
  const int h     = n0 / 192;
  #pragma unroll
  for (int t = 0; t < 4; ++t) {
    const int dk = t * 16 + (lane & 15);
    const float bb = bias[n0 + dk];
    #pragma unroll
    for (int g = 0; g < 8; ++g) {
      const int m = m0 + g + 8 * half;
      const int bidx = m >> 11;          // / L_
      const int l    = m & (L_ - 1);
      const float v  = acc[t][g] + bb;
      if (which == 0)
        q[((size_t)(bidx * H_ + h) * L_ + l) * DK_ + dk] = (f16)(v * 0.125f);
      else if (which == 1)
        k[((size_t)(bidx * H_ + h) * L_ + l) * DK_ + dk] = (f16)v;
      else
        vt[((size_t)(bidx * H_ + h) * DK_ + dk) * L_ + l] = (f16)v;
    }
  }
}

// ---------------------------------------------------------------------------
// Kernel 2: flash attention. One wave per 16-query tile (4 waves / block).
// S^T = K * Q^T so softmax row stats are lane-local; P^T C/D fragment packs
// in-lane into the P A-operand fragment (no cross-lane transpose needed).
// ---------------------------------------------------------------------------
__global__ void __launch_bounds__(128)
attn_kernel(const f16* __restrict__ q, const f16* __restrict__ k,
            const f16* __restrict__ vt, f16* __restrict__ out /*[MTOT,D]*/) {
  const int wave = threadIdx.x >> 5;
  const int lane = threadIdx.x & 31;
  const int h = blockIdx.y, b = blockIdx.z;
  const int bh = b * H_ + h;
  const int q0 = blockIdx.x * 64 + wave * 16;

  const f16* qp = q  + ((size_t)bh * L_ + q0) * DK_;
  const f16* kp = k  + (size_t)bh * L_ * DK_;
  const f16* vp = vt + (size_t)bh * DK_ * L_;

  const v16h qf0 = load_frag(qp, DK_, 0);
  const v16h qf1 = load_frag(qp, DK_, 32);

  v8f zero = {0.f, 0.f, 0.f, 0.f, 0.f, 0.f, 0.f, 0.f};
  v8f o0 = zero, o1 = zero, o2 = zero, o3 = zero;
  float mrow = -3.0e38f;   // running max for q-row (lane&15)
  float ssum = 0.f;        // running denom for q-row (lane&15)
  const int rbase = (lane >> 4) * 8;   // row base for C/D-layout accumulators

  for (int kv = 0; kv < L_; kv += 32) {
    if (kv + 32 < L_)      // warm next K tile (global_prefetch_b8)
      __builtin_prefetch(kp + (size_t)(kv + 32 + (lane & 15)) * DK_, 0, 1);

    // scores^T: two 16(kv) x 16(q) tiles, dk=64 -> 2 wmma each
    v16h k0a = load_frag(kp + (size_t)kv * DK_,        DK_, 0);
    v16h k0b = load_frag(kp + (size_t)kv * DK_,        DK_, 32);
    v16h k1a = load_frag(kp + (size_t)(kv + 16) * DK_, DK_, 0);
    v16h k1b = load_frag(kp + (size_t)(kv + 16) * DK_, DK_, 32);
    v8f st0 = wmma16(k0a, qf0, zero); st0 = wmma16(k0b, qf1, st0);
    v8f st1 = wmma16(k1a, qf0, zero); st1 = wmma16(k1b, qf1, st1);

    // online softmax: per-lane max over its 16 kv entries, merge halves
    float lm = st0[0];
    #pragma unroll
    for (int g = 0; g < 8; ++g) { lm = fmaxf(lm, st0[g]); lm = fmaxf(lm, st1[g]); }
    lm = fmaxf(lm, __shfl_xor(lm, 16, 32));
    const float mnew = fmaxf(mrow, lm);
    const float corr = __expf(mrow - mnew);

    float p0[8], p1[8], psum = 0.f;
    #pragma unroll
    for (int g = 0; g < 8; ++g) {
      p0[g] = __expf(st0[g] - mnew);
      p1[g] = __expf(st1[g] - mnew);
      psum += p0[g] + p1[g];
    }
    psum += __shfl_xor(psum, 16, 32);
    ssum = ssum * corr + psum;
    mrow = mnew;

    // rescale O accumulators: row m = g + rbase, factor lives in lane m&15
    #pragma unroll
    for (int g = 0; g < 8; ++g) {
      const float c = __shfl(corr, rbase + g, 32);
      o0[g] *= c; o1[g] *= c; o2[g] *= c; o3[g] *= c;
    }

    // P^T C/D fragment -> P A-operand fragment: pure in-lane f32->f16 pack
    v16h pa;
    #pragma unroll
    for (int g = 0; g < 8; ++g) { pa[g] = (f16)p0[g]; pa[8 + g] = (f16)p1[g]; }

    // O += P(16x32) @ V(32x64): B-operand = rows of V^T (contiguous in vt)
    v16h v0 = load_frag(vp + (size_t)(0 * 16) * L_, L_, kv);
    v16h v1 = load_frag(vp + (size_t)(1 * 16) * L_, L_, kv);
    v16h v2 = load_frag(vp + (size_t)(2 * 16) * L_, L_, kv);
    v16h v3 = load_frag(vp + (size_t)(3 * 16) * L_, L_, kv);
    o0 = wmma16(pa, v0, o0);
    o1 = wmma16(pa, v1, o1);
    o2 = wmma16(pa, v2, o2);
    o3 = wmma16(pa, v3, o3);
  }

  // normalize and store to [B*L, D] f16 (col = h*64 + dk)
  const float rinv = 1.0f / ssum;
  const int nlo = lane & 15;
  #pragma unroll
  for (int g = 0; g < 8; ++g) {
    const float s = __shfl(rinv, rbase + g, 32);
    const int m = q0 + g + rbase;
    const size_t rowoff = ((size_t)b * L_ + m) * D_ + h * DK_;
    out[rowoff +  0 + nlo] = (f16)(o0[g] * s);
    out[rowoff + 16 + nlo] = (f16)(o1[g] * s);
    out[rowoff + 32 + nlo] = (f16)(o2[g] * s);
    out[rowoff + 48 + nlo] = (f16)(o3[g] * s);
  }
}

// ---------------------------------------------------------------------------
// Kernel 3: output projection  out = A @ Wout + b  (f16 in, f32 out)
// ---------------------------------------------------------------------------
__global__ void __launch_bounds__(128)
out_gemm(const f16* __restrict__ a, const f16* __restrict__ wT,
         const float* __restrict__ bias, float* __restrict__ out) {
  const int wave = threadIdx.x >> 5;
  const int lane = threadIdx.x & 31;
  const int m0 = blockIdx.y * 64 + wave * 16;
  const int n0 = blockIdx.x * 64;

  v8f zero = {0.f, 0.f, 0.f, 0.f, 0.f, 0.f, 0.f, 0.f};
  v8f acc[4] = {zero, zero, zero, zero};

  const f16* ar = a + (size_t)m0 * D_;
  for (int kk = 0; kk < D_; kk += 32) {
    v16h af = load_frag(ar, D_, kk);
    #pragma unroll
    for (int t = 0; t < 4; ++t) {
      v16h bf = load_frag(wT + (size_t)(n0 + t * 16) * D_, D_, kk);
      acc[t] = wmma16(af, bf, acc[t]);
    }
  }

  const int half = lane >> 4;
  #pragma unroll
  for (int t = 0; t < 4; ++t) {
    const int n = n0 + t * 16 + (lane & 15);
    const float bb = bias[n];
    #pragma unroll
    for (int g = 0; g < 8; ++g) {
      const int m = m0 + g + 8 * half;
      out[(size_t)m * D_ + n] = acc[t][g] + bb;
    }
  }
}

// ---------------------------------------------------------------------------
extern "C" void kernel_launch(void* const* d_in, const int* in_sizes, int n_in,
                              void* d_out, int out_size, void* d_ws, size_t ws_size,
                              hipStream_t stream) {
  const float* x     = (const float*)d_in[0];
  const float* w_qkv = (const float*)d_in[1];
  const float* b_qkv = (const float*)d_in[2];
  const float* w_out = (const float*)d_in[3];
  const float* b_out = (const float*)d_in[4];
  float* y = (float*)d_out;

  char* ws = (char*)d_ws;
  f16* wqkvT = (f16*)ws;  ws += (size_t)NQKV * D_ * sizeof(f16);          // 6 MB
  f16* woutT = (f16*)ws;  ws += (size_t)D_ * D_ * sizeof(f16);            // 2 MB
  f16* q     = (f16*)ws;  ws += (size_t)B_ * H_ * L_ * DK_ * sizeof(f16); // 8 MB
  f16* k     = (f16*)ws;  ws += (size_t)B_ * H_ * L_ * DK_ * sizeof(f16); // 8 MB
  f16* vt    = (f16*)ws;  ws += (size_t)B_ * H_ * DK_ * L_ * sizeof(f16); // 8 MB
  f16* attn  = (f16*)ws;                                                  // 8 MB

  dim3 tb(32, 8);
  transpose_f32_to_f16<<<dim3(NQKV / 32, D_ / 32), tb, 0, stream>>>(w_qkv, wqkvT, D_, NQKV);
  transpose_f32_to_f16<<<dim3(D_ / 32, D_ / 32),   tb, 0, stream>>>(w_out, woutT, D_, D_);

  qkv_gemm<<<dim3(NQKV / 64, MTOT / 64), 128, 0, stream>>>(x, wqkvT, b_qkv, q, k, vt);
  attn_kernel<<<dim3(L_ / 64, H_, B_),   128, 0, stream>>>(q, k, vt, attn);
  out_gemm<<<dim3(D_ / 64, MTOT / 64),   128, 0, stream>>>(attn, woutT, b_out, y);
}
